// Attention_19439021982042
// MI455X (gfx1250) — matrix operand order repaired and checked by reading, not compiled, not run on hardware
//
#include <hip/hip_runtime.h>
#include <hip/hip_bf16.h>

typedef __bf16 bf16_t;
typedef __attribute__((ext_vector_type(16))) __bf16 v16bf;
typedef __attribute__((ext_vector_type(8)))  __bf16 v8bf;
typedef __attribute__((ext_vector_type(8)))  float  v8f;

// ---------------------------------------------------------------------------
// WMMA helpers (gfx1250: V_WMMA_F32_16X16X32_BF16, wave32)
// ---------------------------------------------------------------------------
__device__ __forceinline__ v8f v8f_zero() {
  v8f z;
#pragma unroll
  for (int i = 0; i < 8; ++i) z[i] = 0.0f;
  return z;
}

__device__ __forceinline__ v8f wmma_bf16(v16bf a, v16bf b, v8f c) {
  // (neg_a, A, neg_b, B, c_mod, C, reuse_a, reuse_b)
  return __builtin_amdgcn_wmma_f32_16x16x32_bf16(false, a, false, b, (short)0, c,
                                                 false, false);
}

// A-matrix fragment (16x32, 16-bit): lane half 0 -> K {0..7, 16..23},
// lane half 1 -> K {8..15, 24..31}.  rowptr = &A[row][0], row = lane&15.
__device__ __forceinline__ v16bf load_afrag(const bf16_t* rowptr, int k0, int half) {
  const v8bf* p0 = reinterpret_cast<const v8bf*>(rowptr + k0 + half * 8);
  const v8bf* p1 = reinterpret_cast<const v8bf*>(rowptr + k0 + half * 8 + 16);
  v8bf lo = *p0;
  v8bf hi = *p1;
  v16bf r;
#pragma unroll
  for (int i = 0; i < 8; ++i) { r[i] = lo[i]; r[i + 8] = hi[i]; }
  return r;
}

// B-matrix fragment (32x16, 16-bit): lane half 0 -> K {0..7, 16..23},
// lane half 1 -> K {8..15, 24..31}.  colptr = &B^T[col][0] (K contiguous in memory), col = lane&15.
__device__ __forceinline__ v16bf load_bfrag(const bf16_t* colptr, int k0, int half) {
  const v8bf* p0 = reinterpret_cast<const v8bf*>(colptr + k0 + half * 8);
  const v8bf* p1 = reinterpret_cast<const v8bf*>(colptr + k0 + half * 8 + 16);
  v8bf lo = *p0;
  v8bf hi = *p1;
  v16bf r;
#pragma unroll
  for (int i = 0; i < 8; ++i) { r[i] = lo[i]; r[i + 8] = hi[i]; }
  return r;
}

// ---------------------------------------------------------------------------
// Async global->LDS staging of a 32(row) x 32(k) bf16 tile (2 KB).
// 128 lanes x 16 B, one GLOBAL_LOAD_ASYNC_TO_LDS_B128 each (ASYNCcnt path).
// base: bf16 row-major [.,1024]; rows n0..n0+31, k-chunk k0..k0+31.
// ---------------------------------------------------------------------------
__device__ __forceinline__ void stage_b_async(const bf16_t* base, int n0, int k0,
                                              bf16_t* sB /* [32][32] */) {
  const int t = threadIdx.x;
  if (t < 128) {
    const int row = t >> 2;        // 0..31
    const int seg = (t & 3) * 8;   // bf16 element offset within the 32-wide chunk
    unsigned goff  = (unsigned)(((n0 + row) * 1024 + k0 + seg) * 2);  // bytes
    unsigned laddr = (unsigned)(size_t)(sB + row * 32 + seg);         // LDS offset
    asm volatile("global_load_async_to_lds_b128 %0, %1, %2"
                 :: "v"(laddr), "v"(goff), "s"(base) : "memory");
  }
}

__device__ __forceinline__ void wait_async_all() {
  asm volatile("s_wait_asynccnt 0x0" ::: "memory");
}

// ---------------------------------------------------------------------------
// Problem constants
// ---------------------------------------------------------------------------
#define BATCH 4
#define SEQ   2048
#define DMODEL 1024
#define NHEAD 16
#define HDIM  64
#define MROWS (BATCH * SEQ)          // 8192
#define NQKV  (3 * DMODEL)           // 3072

// ---------------------------------------------------------------------------
// f32 -> bf16 conversion kernels
// ---------------------------------------------------------------------------
__global__ void cvt_bf16(const float* __restrict__ src, bf16_t* __restrict__ dst, int n) {
  int i = blockIdx.x * blockDim.x + threadIdx.x;
  if (i < n) dst[i] = (bf16_t)src[i];
}

// dst[c][r] = src[r][c]   (transpose w_proj so B-fragments read contiguous K)
__global__ void cvt_bf16_T(const float* __restrict__ src, bf16_t* __restrict__ dst,
                           int rows, int cols) {
  int i = blockIdx.x * blockDim.x + threadIdx.x;
  if (i < rows * cols) {
    int r = i / cols, c = i % cols;
    dst[(size_t)c * rows + r] = (bf16_t)src[i];
  }
}

// ---------------------------------------------------------------------------
// QKV GEMM: qkv[m][n] = sum_k xb[m][k] * w_attn[n][k] + b_attn[n]
// Block = 8 waves; wave computes 64x32; B tile double-buffered through LDS
// via GLOBAL_LOAD_ASYNC_TO_LDS_B128.  grid (M/512, 3072/32).
// Epilogue scatters q/k/v into head layouts; k,v also to fp32 `present`.
// ---------------------------------------------------------------------------
__global__ __launch_bounds__(256) void qkv_wmma(
    const bf16_t* __restrict__ xb, const bf16_t* __restrict__ wb,
    const float* __restrict__ bias,
    bf16_t* __restrict__ qh, bf16_t* __restrict__ kh, bf16_t* __restrict__ vT,
    float* __restrict__ presK, float* __restrict__ presV) {
  __shared__ bf16_t sB[2][32 * 32];

  const int lane = threadIdx.x & 31, wave = threadIdx.x >> 5;
  const int half = lane >> 4, l15 = lane & 15;
  const int m0 = blockIdx.x * 512 + wave * 64;
  const int n0 = blockIdx.y * 32;

  v8f acc[4][2];
#pragma unroll
  for (int i = 0; i < 4; ++i)
#pragma unroll
    for (int j = 0; j < 2; ++j) acc[i][j] = v8f_zero();

  const bf16_t* arow[4];
#pragma unroll
  for (int tm = 0; tm < 4; ++tm)
    arow[tm] = xb + (size_t)(m0 + tm * 16 + l15) * DMODEL;

  stage_b_async(wb, n0, 0, &sB[0][0]);

  for (int k0 = 0; k0 < DMODEL; k0 += 32) {
    const int buf = (k0 >> 5) & 1;
    wait_async_all();
    __syncthreads();
    if (k0 + 32 < DMODEL) stage_b_async(wb, n0, k0 + 32, &sB[buf ^ 1][0]);

    v16bf b0 = load_bfrag(&sB[buf][(0 + l15) * 32], 0, half);
    v16bf b1 = load_bfrag(&sB[buf][(16 + l15) * 32], 0, half);
#pragma unroll
    for (int tm = 0; tm < 4; ++tm) {
      __builtin_prefetch(arow[tm] + k0 + 64, 0, 0);   // global_prefetch_b8
      v16bf a = load_afrag(arow[tm], k0, half);
      acc[tm][0] = wmma_bf16(a, b0, acc[tm][0]);
      acc[tm][1] = wmma_bf16(a, b1, acc[tm][1]);
    }
    __syncthreads();
  }

#pragma unroll
  for (int tm = 0; tm < 4; ++tm)
#pragma unroll
    for (int tn = 0; tn < 2; ++tn)
#pragma unroll
      for (int e = 0; e < 8; ++e) {
        int m = m0 + tm * 16 + e + half * 8;
        int n = n0 + tn * 16 + l15;
        float v = acc[tm][tn][e] + bias[n];
        int b = m >> 11, s = m & (SEQ - 1);
        if (n < DMODEL) {                         // Q
          int h = n >> 6, d = n & 63;
          qh[((size_t)(b * NHEAD + h) * SEQ + s) * HDIM + d] = (bf16_t)v;
        } else if (n < 2 * DMODEL) {              // K
          int nk = n - DMODEL;
          int h = nk >> 6, d = nk & 63;
          size_t idx = ((size_t)(b * NHEAD + h) * SEQ + s) * HDIM + d;
          kh[idx] = (bf16_t)v;
          presK[idx] = v;
        } else {                                  // V
          int nv = n - 2 * DMODEL;
          int h = nv >> 6, d = nv & 63;
          size_t idx = ((size_t)(b * NHEAD + h) * SEQ + s) * HDIM + d;
          vT[((size_t)(b * NHEAD + h) * HDIM + d) * SEQ + s] = (bf16_t)v;
          presV[idx] = v;
        }
      }
}

// ---------------------------------------------------------------------------
// Flash attention: one wave per 32 query rows of one (b,h).
// Streams 32-key causal tiles; scores + P@V on WMMA; online softmax via an
// LDS transpose (row-per-lane) + __shfl broadcast of rescale factors.
// Waves have divergent causal trip counts -> no block barriers; each wave
// owns a private LDS slice.  grid (B*H, SEQ/128), block 128.
// ---------------------------------------------------------------------------
#define ATT_WAVES 4
__global__ __launch_bounds__(128) void attn_wmma(
    const bf16_t* __restrict__ qh, const bf16_t* __restrict__ kh,
    const bf16_t* __restrict__ vT, bf16_t* __restrict__ ao) {
  __shared__ float  sS[ATT_WAVES][32][32];
  __shared__ bf16_t sP[ATT_WAVES][32][32];

  const int lane = threadIdx.x & 31, wave = threadIdx.x >> 5;
  const int half = lane >> 4, l15 = lane & 15;
  const int bh = blockIdx.x;
  const int q0 = blockIdx.y * (ATT_WAVES * 32) + wave * 32;

  const bf16_t* Qb = qh + (size_t)bh * SEQ * HDIM;
  const bf16_t* Kb = kh + (size_t)bh * SEQ * HDIM;
  const bf16_t* Vb = vT + (size_t)bh * HDIM * SEQ;

  // Q fragments: [tm][kc] covering rows q0+tm*16.., K-chunks of hd
  v16bf qf[2][2];
#pragma unroll
  for (int tm = 0; tm < 2; ++tm)
#pragma unroll
    for (int kc = 0; kc < 2; ++kc)
      qf[tm][kc] = load_afrag(Qb + (size_t)(q0 + tm * 16 + l15) * HDIM, kc * 32, half);

  v8f acc[2][4];
#pragma unroll
  for (int tm = 0; tm < 2; ++tm)
#pragma unroll
    for (int nc = 0; nc < 4; ++nc) acc[tm][nc] = v8f_zero();

  float m_i = -3.0e38f;   // running row max (row = lane)
  float l_i = 0.0f;       // running row sum

  for (int kt = 0; kt <= q0 + 31; kt += 32) {
    // ---- scores S = (Q K^T) * 1/sqrt(hd), causal-masked, to LDS ----
#pragma unroll
    for (int tn = 0; tn < 2; ++tn) {
      const bf16_t* kcol = Kb + (size_t)(kt + tn * 16 + l15) * HDIM;
      v16bf b0 = load_bfrag(kcol, 0, half);
      v16bf b1 = load_bfrag(kcol, 32, half);
#pragma unroll
      for (int tm = 0; tm < 2; ++tm) {
        v8f s = wmma_bf16(qf[tm][0], b0, v8f_zero());
        s = wmma_bf16(qf[tm][1], b1, s);
#pragma unroll
        for (int e = 0; e < 8; ++e) {
          int r = tm * 16 + e + half * 8;
          int c = tn * 16 + l15;
          float val = s[e] * 0.125f;                 // 1/sqrt(64)
          if (kt + c > q0 + r) val = -3.0e38f;       // causal mask
          sS[wave][r][c] = val;
        }
      }
    }

    // ---- per-row online softmax (row = lane) ----
    const float* srow = &sS[wave][lane][0];
    float rmax = -3.0e38f;
#pragma unroll
    for (int j = 0; j < 32; ++j) rmax = fmaxf(rmax, srow[j]);
    float m_new = fmaxf(m_i, rmax);
    float alpha = __expf(m_i - m_new);
    float psum = 0.0f;
#pragma unroll
    for (int j = 0; j < 32; ++j) {
      float pv = __expf(srow[j] - m_new);
      psum += pv;
      sP[wave][lane][j] = (bf16_t)pv;
    }
    l_i = l_i * alpha + psum;
    m_i = m_new;

    // ---- rescale accumulator rows by alpha (broadcast via shuffle) ----
#pragma unroll
    for (int tm = 0; tm < 2; ++tm)
#pragma unroll
      for (int e = 0; e < 8; ++e) {
        int srcLane = tm * 16 + e + half * 8;
        float av = __shfl(alpha, srcLane, 32);
#pragma unroll
        for (int nc = 0; nc < 4; ++nc) acc[tm][nc][e] *= av;
      }

    // ---- P fragments from LDS, V fragments from transposed global V ----
    v16bf pf[2];
#pragma unroll
    for (int tm = 0; tm < 2; ++tm)
      pf[tm] = load_afrag(&sP[wave][tm * 16 + l15][0], 0, half);

#pragma unroll
    for (int nc = 0; nc < 4; ++nc) {
      v16bf vf = load_bfrag(Vb + (size_t)(nc * 16 + l15) * SEQ, kt, half);
      acc[0][nc] = wmma_bf16(pf[0], vf, acc[0][nc]);
      acc[1][nc] = wmma_bf16(pf[1], vf, acc[1][nc]);
    }
  }

  // ---- finalize: divide by row sum, store merged-head bf16 activations ----
  float linv = 1.0f / l_i;
  int b = bh >> 4, h = bh & (NHEAD - 1);
#pragma unroll
  for (int tm = 0; tm < 2; ++tm)
#pragma unroll
    for (int e = 0; e < 8; ++e) {
      int r = tm * 16 + e + half * 8;
      float lv = __shfl(linv, r, 32);
      int s = q0 + r;
      bf16_t* orow = ao + ((size_t)b * SEQ + s) * DMODEL + h * HDIM;
#pragma unroll
      for (int nc = 0; nc < 4; ++nc) orow[nc * 16 + l15] = (bf16_t)(acc[tm][nc][e] * lv);
    }
}

// ---------------------------------------------------------------------------
// Proj GEMM: out[m][n] = sum_k a[m][k] * w_proj[k][n] + b_proj[n]
// Same structure as qkv_wmma (64x32 per wave, async-LDS B double buffering);
// wpT is w_proj transposed to [n][k] bf16 so staging reads contiguous K.
// ---------------------------------------------------------------------------
__global__ __launch_bounds__(256) void proj_wmma(
    const bf16_t* __restrict__ ab, const bf16_t* __restrict__ wpT,
    const float* __restrict__ bias, float* __restrict__ out) {
  __shared__ bf16_t sB[2][32 * 32];

  const int lane = threadIdx.x & 31, wave = threadIdx.x >> 5;
  const int half = lane >> 4, l15 = lane & 15;
  const int m0 = blockIdx.x * 512 + wave * 64;
  const int n0 = blockIdx.y * 32;

  v8f acc[4][2];
#pragma unroll
  for (int i = 0; i < 4; ++i)
#pragma unroll
    for (int j = 0; j < 2; ++j) acc[i][j] = v8f_zero();

  const bf16_t* arow[4];
#pragma unroll
  for (int tm = 0; tm < 4; ++tm)
    arow[tm] = ab + (size_t)(m0 + tm * 16 + l15) * DMODEL;

  stage_b_async(wpT, n0, 0, &sB[0][0]);

  for (int k0 = 0; k0 < DMODEL; k0 += 32) {
    const int buf = (k0 >> 5) & 1;
    wait_async_all();
    __syncthreads();
    if (k0 + 32 < DMODEL) stage_b_async(wpT, n0, k0 + 32, &sB[buf ^ 1][0]);

    v16bf b0 = load_bfrag(&sB[buf][(0 + l15) * 32], 0, half);
    v16bf b1 = load_bfrag(&sB[buf][(16 + l15) * 32], 0, half);
#pragma unroll
    for (int tm = 0; tm < 4; ++tm) {
      __builtin_prefetch(arow[tm] + k0 + 64, 0, 0);
      v16bf a = load_afrag(arow[tm], k0, half);
      acc[tm][0] = wmma_bf16(a, b0, acc[tm][0]);
      acc[tm][1] = wmma_bf16(a, b1, acc[tm][1]);
    }
    __syncthreads();
  }

#pragma unroll
  for (int tm = 0; tm < 4; ++tm)
#pragma unroll
    for (int tn = 0; tn < 2; ++tn)
#pragma unroll
      for (int e = 0; e < 8; ++e) {
        int m = m0 + tm * 16 + e + half * 8;
        int n = n0 + tn * 16 + l15;
        out[(size_t)m * DMODEL + n] = acc[tm][tn][e] + bias[n];
      }
}

// ---------------------------------------------------------------------------
// Host-side launch
// ---------------------------------------------------------------------------
extern "C" void kernel_launch(void* const* d_in, const int* in_sizes, int n_in,
                              void* d_out, int out_size, void* d_ws, size_t ws_size,
                              hipStream_t stream) {
  const float* x      = (const float*)d_in[0];  // [4,2048,1024]
  const float* w_attn = (const float*)d_in[1];  // [3072,1024]
  const float* b_attn = (const float*)d_in[2];  // [3072]
  const float* w_proj = (const float*)d_in[3];  // [1024,1024]
  const float* b_proj = (const float*)d_in[4];  // [1024]

  float* out = (float*)d_out;                              // [4,2048,1024]
  float* presK = out + (size_t)MROWS * DMODEL;             // present[0]: [B,H,S,hd]
  float* presV = presK + (size_t)BATCH * NHEAD * SEQ * HDIM;

  // workspace carve-up (bf16 intermediates; ~92 MB total, L2-resident)
  char* ws = (char*)d_ws;
  bf16_t* xb  = (bf16_t*)ws; ws += (size_t)MROWS * DMODEL * 2;        // 16 MB
  bf16_t* wb  = (bf16_t*)ws; ws += (size_t)NQKV * DMODEL * 2;         //  6 MB
  bf16_t* qh  = (bf16_t*)ws; ws += (size_t)BATCH * NHEAD * SEQ * HDIM * 2;
  bf16_t* kh  = (bf16_t*)ws; ws += (size_t)BATCH * NHEAD * SEQ * HDIM * 2;
  bf16_t* vT  = (bf16_t*)ws; ws += (size_t)BATCH * NHEAD * HDIM * SEQ * 2;
  bf16_t* ao  = (bf16_t*)ws; ws += (size_t)MROWS * DMODEL * 2;
  bf16_t* wpT = (bf16_t*)ws; ws += (size_t)DMODEL * DMODEL * 2;

  const int nx = MROWS * DMODEL;      // 8388608
  const int nw = NQKV * DMODEL;       // 3145728
  const int np = DMODEL * DMODEL;     // 1048576

  cvt_bf16<<<(nx + 255) / 256, 256, 0, stream>>>(x, xb, nx);
  cvt_bf16<<<(nw + 255) / 256, 256, 0, stream>>>(w_attn, wb, nw);
  cvt_bf16_T<<<(np + 255) / 256, 256, 0, stream>>>(w_proj, wpT, DMODEL, DMODEL);

  qkv_wmma<<<dim3(MROWS / 512, NQKV / 32), 256, 0, stream>>>(
      xb, wb, b_attn, qh, kh, vT, presK, presV);

  attn_wmma<<<dim3(BATCH * NHEAD, SEQ / (ATT_WAVES * 32)), ATT_WAVES * 32, 0, stream>>>(
      qh, kh, vT, ao);

  proj_wmma<<<dim3(MROWS / 512, DMODEL / 32), 256, 0, stream>>>(ao, wpT, b_proj, out);
}
